// Charge_Fusion_71760313581913
// MI455X (gfx1250) — matrix operand order repaired and checked by reading, not compiled, not run on hardware
//
#include <hip/hip_runtime.h>
#include <hip/hip_bf16.h>
#include <math.h>

#define HID  768
#define NLAB 200
#define SEQ  512
#define QN   256

typedef __attribute__((ext_vector_type(16))) __bf16 v16bf;
typedef __attribute__((ext_vector_type(8)))  float  v8f;
typedef __attribute__((ext_vector_type(4)))  int    v4i;

static __device__ inline v8f vzero8() {
  v8f z = {0.f, 0.f, 0.f, 0.f, 0.f, 0.f, 0.f, 0.f};
  return z;
}

static __device__ inline v8f wmma_bf16(v16bf a, v16bf b, v8f c) {
  // D = A(16x32 bf16) x B(32x16 bf16) + C(16x16 f32)
  return __builtin_amdgcn_wmma_f32_16x16x32_bf16(
      false, a, false, b, (short)0, c, false, false);
}

// A fragment (16x32) from bf16 row-major [16][lda]; two 16B vector loads.
// Element i of lane L: M = L&15; K = (i<8) ? half*8+i : 16+half*8+(i-8).
static __device__ inline v16bf load_a_bf16(const __bf16* base, int lda, int kbase, int lane) {
  int m = lane & 15, h = lane >> 4;
  const __bf16* row = base + (size_t)m * lda + kbase;
  union { v4i q[2]; v16bf v; } u;
  u.q[0] = *(const v4i*)(row + h * 8);
  u.q[1] = *(const v4i*)(row + 16 + h * 8);
  return u.v;
}

// A fragment from f32 source (attn probabilities), converted on the fly.
static __device__ inline v16bf load_a_f32(const float* base, int lda, int kbase, int lane) {
  int m = lane & 15, h = lane >> 4;
  const float* row = base + (size_t)m * lda + kbase;
  v16bf a;
#pragma unroll
  for (int i = 0; i < 8; ++i) a[i] = (__bf16)row[h * 8 + i];
#pragma unroll
  for (int i = 0; i < 8; ++i) a[8 + i] = (__bf16)row[16 + h * 8 + i];
  return a;
}

// B fragment (32x16) with B[k][n] = src[n][k] (src row-major [N][ldb]).
// Per lane: K = half*8+i (i<8), 16+half*8+(i-8) -> two 16B vector loads.
static __device__ inline v16bf load_bt_bf16(const __bf16* src, int ldb, int nbase, int kbase, int lane) {
  const __bf16* row = src + (size_t)(nbase + (lane & 15)) * ldb + kbase + (lane >> 4) * 8;
  union { v4i q[2]; v16bf v; } u;
  u.q[0] = *(const v4i*)(row);
  u.q[1] = *(const v4i*)(row + 16);
  return u.v;
}

// CDNA5 async global->LDS copy (ASYNCcnt-tracked), 16 bytes per lane.
// VDST = LDS byte offset VGPR, VADDR = 64-bit global address pair.
static __device__ inline void async_copy_b128(const void* gsrc, void* ldst) {
  unsigned loff = (unsigned)(size_t)ldst;               // low 32 bits = LDS addr
  unsigned long long ga = (unsigned long long)(size_t)gsrc;
  asm volatile("global_load_async_to_lds_b128 %0, %1, off"
               :: "v"(loff), "v"(ga) : "memory");
}
static __device__ inline void async_wait0() {
  asm volatile("s_wait_asynccnt 0x0" ::: "memory");
}

// ---------------------------------------------------------------------------
// out[m][n] = X[m][:] . W[n][:] + bias[n] (bf16 W, bf16 out).
// One block per 16-row m-tile (X read exactly once from HBM); 4 waves,
// each wave computes 16x64 macro-tiles (4 accumulators reuse one A fragment).
// Optionally writes h-major transposed copy outT[cidx][n][s].
// ---------------------------------------------------------------------------
__global__ __launch_bounds__(128) void proj_kernel(
    const float* __restrict__ X, const __bf16* __restrict__ Wbf,
    const float* __restrict__ bias, __bf16* __restrict__ outA,
    __bf16* __restrict__ outT) {
  __shared__ __align__(16) __bf16 sX[16 * HID];  // 24 KB bf16 A tile

  int tid = threadIdx.x, lane = tid & 31, wave = tid >> 5;
  int mbase = blockIdx.x * 16;

  for (int idx = tid; idx < 16 * HID; idx += 128) {
    int m = idx / HID, k = idx % HID;
    sX[idx] = (__bf16)X[(size_t)(mbase + m) * HID + k];
  }
  __syncthreads();

  int ln = lane & 15, h2 = lane >> 4;
  for (int g = wave; g < HID / 64; g += 4) {
    int nb = g * 64;
    v8f a0 = vzero8(), a1 = vzero8(), a2 = vzero8(), a3 = vzero8();
    for (int kb = 0; kb < HID; kb += 32) {
      v16bf a = load_a_bf16(sX, HID, kb, lane);
      a0 = wmma_bf16(a, load_bt_bf16(Wbf, HID, nb,      kb, lane), a0);
      a1 = wmma_bf16(a, load_bt_bf16(Wbf, HID, nb + 16, kb, lane), a1);
      a2 = wmma_bf16(a, load_bt_bf16(Wbf, HID, nb + 32, kb, lane), a2);
      a3 = wmma_bf16(a, load_bt_bf16(Wbf, HID, nb + 48, kb, lane), a3);
    }
    v8f accs[4] = {a0, a1, a2, a3};
#pragma unroll
    for (int t = 0; t < 4; ++t) {
      int n = nb + t * 16 + ln;
      float bn = bias[n];
#pragma unroll
      for (int j = 0; j < 8; ++j) {
        int m = mbase + j + 8 * h2;
        float v = accs[t][j] + bn;
        outA[(size_t)m * HID + n] = (__bf16)v;
        if (outT) {
          int cidx = m / SEQ, s = m % SEQ;
          outT[((size_t)cidx * HID + n) * SEQ + s] = (__bf16)v;
        }
      }
    }
  }
}

__global__ void cvt_bf16_kernel(const float* __restrict__ in,
                                __bf16* __restrict__ out, int n) {
  int i = blockIdx.x * blockDim.x + threadIdx.x;
  if (i < n) out[i] = (__bf16)in[i];
}

__global__ void bias_sum_kernel(const float* __restrict__ bias,
                                float* __restrict__ out) {
  __shared__ float red[256];
  int t = threadIdx.x;
  float s = 0.f;
  for (int i = t; i < HID; i += 256) s += bias[i];
  red[t] = s;
  __syncthreads();
  for (int o = 128; o > 0; o >>= 1) {
    if (t < o) red[t] += red[t + o];
    __syncthreads();
  }
  if (t == 0) out[0] = red[0];
}

// ---------------------------------------------------------------------------
// One workgroup per (16-row q-tile, label c); grid = (QN/16, NLAB) so the 16
// blocks sharing a 1.5MB label tile run adjacently (L2-resident reuse).
// Phase 1: scores = qtile . c^T  (WMMA, 16x64 macro-tiles)
// Phase 2: masked softmax over s
// Phase 3: emb = attn . c ; sQ <- bf16(q + emb) in place
// Phase 4: tanh(fused . Wf^T + b_fusion) . Ws[c] + sum(bias)
// ---------------------------------------------------------------------------
__global__ __launch_bounds__(128) void attn_fuse_kernel(
    const __bf16* __restrict__ q_bf, const __bf16* __restrict__ c_bf,
    const __bf16* __restrict__ cT_bf, const __bf16* __restrict__ Wf_bf,
    const float* __restrict__ mask, const float* __restrict__ b_fusion,
    const float* __restrict__ Ws, const float* __restrict__ bias_sum,
    float* __restrict__ out) {
  __shared__ __align__(16) __bf16 sQ[16 * HID];  // 24 KB: q tile -> fused tile
  __shared__ __align__(16) float  sS[16 * SEQ];  // 32 KB: scores -> attn
  __shared__ float sRed[16 * 8];
  __shared__ float sOut[16];

  int qbase = blockIdx.x * 16;
  int c = blockIdx.y;
  int tid = threadIdx.x, lane = tid & 31, wave = tid >> 5;
  int half = lane >> 4, ln = lane & 15;

  // Async-stage the q tile (ASYNCcnt path): 1536 x 16B chunks over 128 threads.
  const __bf16* qsrc = q_bf + (size_t)qbase * HID;
  for (int t = tid; t < (16 * HID) / 8; t += 128)
    async_copy_b128(qsrc + t * 8, &sQ[t * 8]);
  if (tid < 16) sOut[tid] = 0.0f;
  async_wait0();
  __syncthreads();

  // ---- Phase 1: scores ----
  const __bf16* crow = c_bf + (size_t)c * SEQ * HID;
  for (int g = wave; g < SEQ / 64; g += 4) {
    int sb = g * 64;
    if (g + 4 < SEQ / 64)
      __builtin_prefetch(crow + (size_t)(g + 4) * 64 * HID, 0, 1);
    v8f a0 = vzero8(), a1 = vzero8(), a2 = vzero8(), a3 = vzero8();
    for (int kb = 0; kb < HID; kb += 32) {
      v16bf a = load_a_bf16(sQ, HID, kb, lane);
      a0 = wmma_bf16(a, load_bt_bf16(crow, HID, sb,      kb, lane), a0);
      a1 = wmma_bf16(a, load_bt_bf16(crow, HID, sb + 16, kb, lane), a1);
      a2 = wmma_bf16(a, load_bt_bf16(crow, HID, sb + 32, kb, lane), a2);
      a3 = wmma_bf16(a, load_bt_bf16(crow, HID, sb + 48, kb, lane), a3);
    }
    v8f accs[4] = {a0, a1, a2, a3};
#pragma unroll
    for (int t = 0; t < 4; ++t)
#pragma unroll
      for (int j = 0; j < 8; ++j)
        sS[(j + 8 * half) * SEQ + sb + t * 16 + ln] = accs[t][j];
  }
  __syncthreads();

  // ---- Phase 2: masked softmax over s (8 threads per row) ----
  {
    int r = tid & 15, g = tid >> 4;
    const float* mrow = mask + (size_t)c * SEQ;
    float mx = -3.0e38f;
    for (int s = g; s < SEQ; s += 8) {
      float v = sS[r * SEQ + s] + (1.0f - mrow[s]) * -1.0e9f;
      sS[r * SEQ + s] = v;
      mx = fmaxf(mx, v);
    }
    sRed[r * 8 + g] = mx;
    __syncthreads();
    if (g == 0) {
      float m2 = sRed[r * 8];
      for (int i = 1; i < 8; ++i) m2 = fmaxf(m2, sRed[r * 8 + i]);
      sRed[r * 8] = m2;
    }
    __syncthreads();
    float rowmax = sRed[r * 8];
    float sum = 0.f;
    for (int s = g; s < SEQ; s += 8) {
      float e = __expf(sS[r * SEQ + s] - rowmax);
      sS[r * SEQ + s] = e;
      sum += e;
    }
    __syncthreads();
    sRed[r * 8 + g] = sum;
    __syncthreads();
    if (g == 0) {
      float s2 = sRed[r * 8];
      for (int i = 1; i < 8; ++i) s2 += sRed[r * 8 + i];
      sRed[r * 8] = s2;
    }
    __syncthreads();
    float inv = 1.0f / sRed[r * 8];
    for (int s = g; s < SEQ; s += 8) sS[r * SEQ + s] *= inv;
  }
  __syncthreads();

  // ---- Phase 3: emb = attn . c ; fused = q + emb (in place, bf16) ----
  const __bf16* cTrow = cT_bf + (size_t)c * HID * SEQ;
  for (int g = wave; g < HID / 64; g += 4) {
    int nb = g * 64;
    v8f a0 = vzero8(), a1 = vzero8(), a2 = vzero8(), a3 = vzero8();
    for (int kb = 0; kb < SEQ; kb += 32) {
      v16bf a = load_a_f32(sS, SEQ, kb, lane);
      a0 = wmma_bf16(a, load_bt_bf16(cTrow, SEQ, nb,      kb, lane), a0);
      a1 = wmma_bf16(a, load_bt_bf16(cTrow, SEQ, nb + 16, kb, lane), a1);
      a2 = wmma_bf16(a, load_bt_bf16(cTrow, SEQ, nb + 32, kb, lane), a2);
      a3 = wmma_bf16(a, load_bt_bf16(cTrow, SEQ, nb + 48, kb, lane), a3);
    }
    v8f accs[4] = {a0, a1, a2, a3};
#pragma unroll
    for (int t = 0; t < 4; ++t) {
      int n = nb + t * 16 + ln;
#pragma unroll
      for (int j = 0; j < 8; ++j) {
        int m = j + 8 * half;
        float f = (float)sQ[m * HID + n] + accs[t][j];
        sQ[m * HID + n] = (__bf16)f;
      }
    }
  }
  __syncthreads();

  // ---- Phase 4: fusion GEMM + tanh + Ws dot + row reduction ----
  float ps[8];
#pragma unroll
  for (int j = 0; j < 8; ++j) ps[j] = 0.f;
  for (int g = wave; g < HID / 64; g += 4) {
    int nb = g * 64;
    v8f a0 = vzero8(), a1 = vzero8(), a2 = vzero8(), a3 = vzero8();
    for (int kb = 0; kb < HID; kb += 32) {
      v16bf a = load_a_bf16(sQ, HID, kb, lane);
      a0 = wmma_bf16(a, load_bt_bf16(Wf_bf, HID, nb,      kb, lane), a0);
      a1 = wmma_bf16(a, load_bt_bf16(Wf_bf, HID, nb + 16, kb, lane), a1);
      a2 = wmma_bf16(a, load_bt_bf16(Wf_bf, HID, nb + 32, kb, lane), a2);
      a3 = wmma_bf16(a, load_bt_bf16(Wf_bf, HID, nb + 48, kb, lane), a3);
    }
    v8f accs[4] = {a0, a1, a2, a3};
#pragma unroll
    for (int t = 0; t < 4; ++t) {
      int n = nb + t * 16 + ln;
      float w = Ws[(size_t)c * HID + n];
      float bf = b_fusion[n];
#pragma unroll
      for (int j = 0; j < 8; ++j)
        ps[j] += tanhf(accs[t][j] + bf) * w;
    }
  }
#pragma unroll
  for (int j = 0; j < 8; ++j)
    atomicAdd(&sOut[j + 8 * half], ps[j]);
  __syncthreads();
  if (tid < 16)
    out[(size_t)(qbase + tid) * NLAB + c] = sOut[tid] + bias_sum[0];
}

// ---------------------------------------------------------------------------
extern "C" void kernel_launch(void* const* d_in, const int* in_sizes, int n_in,
                              void* d_out, int out_size, void* d_ws, size_t ws_size,
                              hipStream_t stream) {
  (void)in_sizes; (void)n_in; (void)out_size; (void)ws_size;
  const float* Q_fact   = (const float*)d_in[0];
  const float* charge   = (const float*)d_in[1];
  const float* mask     = (const float*)d_in[2];
  const float* W_fact   = (const float*)d_in[3];
  const float* b_fact   = (const float*)d_in[4];
  const float* W_charge = (const float*)d_in[5];
  const float* b_charge = (const float*)d_in[6];
  const float* W_fusion = (const float*)d_in[7];
  const float* b_fusion = (const float*)d_in[8];
  const float* Ws       = (const float*)d_in[9];
  const float* bias     = (const float*)d_in[10];
  float* out = (float*)d_out;

  char* ws = (char*)d_ws;
  size_t off = 0;
  auto carve = [&](size_t bytes) -> char* {
    char* p = ws + off;
    off = (off + bytes + 255) & ~(size_t)255;
    return p;
  };
  __bf16* q_bf   = (__bf16*)carve((size_t)QN * HID * 2);
  __bf16* c_bf   = (__bf16*)carve((size_t)NLAB * SEQ * HID * 2);
  __bf16* cT_bf  = (__bf16*)carve((size_t)NLAB * HID * SEQ * 2);
  __bf16* Wf_bf  = (__bf16*)carve((size_t)HID * HID * 2);
  __bf16* Wfa_bf = (__bf16*)carve((size_t)HID * HID * 2);
  __bf16* Wch_bf = (__bf16*)carve((size_t)HID * HID * 2);
  float*  bsum   = (float*)carve(sizeof(float));

  // Pre-convert weight matrices to bf16 (L2-resident, re-read from L2).
  int nw = HID * HID;
  cvt_bf16_kernel<<<(nw + 255) / 256, 256, 0, stream>>>(W_fact,   Wfa_bf, nw);
  cvt_bf16_kernel<<<(nw + 255) / 256, 256, 0, stream>>>(W_charge, Wch_bf, nw);
  cvt_bf16_kernel<<<(nw + 255) / 256, 256, 0, stream>>>(W_fusion, Wf_bf,  nw);
  bias_sum_kernel<<<1, 256, 0, stream>>>(bias, bsum);

  // q = Q_fact @ W_fact^T + b_fact  -> bf16
  proj_kernel<<<dim3(QN / 16), 128, 0, stream>>>(
      Q_fact, Wfa_bf, b_fact, q_bf, nullptr);
  // c = charge @ W_charge^T + b_charge -> bf16 [C,S,H] and transposed [C,H,S]
  proj_kernel<<<dim3((NLAB * SEQ) / 16), 128, 0, stream>>>(
      charge, Wch_bf, b_charge, c_bf, cT_bf);

  // attention + fusion + reduction
  attn_fuse_kernel<<<dim3(QN / 16, NLAB), 128, 0, stream>>>(
      q_bf, c_bf, cT_bf, Wf_bf, mask, b_fusion, Ws, bsum, out);
}
